// ParentFeedingLSTMDecoder_18124761989453
// MI455X (gfx1250) — matrix operand order repaired and checked by reading, not compiled, not run on hardware
//
#include <hip/hip_runtime.h>
#include <cstdint>
#include <cstddef>

typedef __attribute__((ext_vector_type(16))) __bf16 v16bf;
typedef __attribute__((ext_vector_type(8)))  __bf16 v8bf;
typedef __attribute__((ext_vector_type(8)))  float  v8f;

#define B_     256
#define S_     128
#define C_     64
#define T_     60
#define DENC   512
#define DRNN   512
#define DATT   512
#define DACT   128
#define DTYP   64
#define NPROD_ 97
#define XIN    1344          // ACT + ATT + ACT + TYP + RNN
#define KCAT   1856          // XIN + RNN
#define NGATE  2048
#define NEGV   (-1e9f)

static __device__ __forceinline__ __bf16 f2bf(float f) {
  union { float f; uint32_t u; } v; v.f = f;
  uint32_t r = v.u + 0x7FFFu + ((v.u >> 16) & 1u);   // round-to-nearest-even
  uint16_t h = (uint16_t)(r >> 16);
  return __builtin_bit_cast(__bf16, h);
}

// ---------------------------------------------------------------------------
// Generic bf16 WMMA GEMM: C[M,N] = A[M,K] * BT[N,K]^T (+bias) (opt tanh)
// A row-major bf16, BT is B pre-transposed (row n holds column n of B).
// Block = 128 threads = 4 waves; wave w -> rows [blockIdx.y*64 + w*16, +16),
// cols [blockIdx.x*64, +64) as 4 accumulator tiles. M%64==0, N%64==0, K%32==0.
// ---------------------------------------------------------------------------
__global__ void wmma_gemm(const __bf16* __restrict__ A, const __bf16* __restrict__ BT,
                          const float* __restrict__ bias,
                          float* __restrict__ Cf, __bf16* __restrict__ Cb,
                          int M, int N, int K, int act)
{
  const int lane = threadIdx.x & 31;
  const int wave = threadIdx.x >> 5;
  const int l    = lane & 15;
  const int half = lane >> 4;
  const int mBase = (blockIdx.y * 4 + wave) * 16;
  const int nBase = blockIdx.x * 64;

  v8f acc[4] = {};
  // A fragment (16x32 bf16): lanes 0-15 row M=l hold K 0..15;
  // lanes 16-31 hold K 16..31  -> base = half*16, second chunk +8.
  const __bf16* pA = A  + (size_t)(mBase + l) * K + (half << 4);
  // B fragment (32x16 bf16): lane col N=l, lanes 0-15 K 0..15, lanes 16-31 K 16..31.
  const __bf16* pB = BT + (size_t)(nBase + l) * K + (half << 4);
  const size_t strB = (size_t)16 * K;

  for (int k0 = 0; k0 < K; k0 += 32) {
    v16bf a;
    {
      v8bf a0 = *(const v8bf*)(pA + k0);
      v8bf a1 = *(const v8bf*)(pA + k0 + 8);
#pragma unroll
      for (int i = 0; i < 8; ++i) { a[i] = a0[i]; a[i + 8] = a1[i]; }
    }
#pragma unroll
    for (int nt = 0; nt < 4; ++nt) {
      v16bf b = *(const v16bf*)(pB + (size_t)nt * strB + k0);
      acc[nt] = __builtin_amdgcn_wmma_f32_16x16x32_bf16(
          false, a, false, b, (short)0, acc[nt], false, false);
    }
  }
  // C/D layout: lane l (0-15) = col, VGPR v = row half*8+v.
#pragma unroll
  for (int nt = 0; nt < 4; ++nt) {
    int col = nBase + nt * 16 + l;
    float bv = bias ? bias[col] : 0.0f;
#pragma unroll
    for (int v = 0; v < 8; ++v) {
      int row = mBase + (half << 3) + v;
      float val = acc[nt][v] + bv;
      if (act) val = tanhf(val);
      if (Cf) Cf[(size_t)row * N + col] = val;
      if (Cb) Cb[(size_t)row * N + col] = f2bf(val);
    }
  }
}

// --- weight prep: f32 [K,N] -> bf16 [N,K] (transposed for B-fragment loads) --
__global__ void transpose_bf16(const float* __restrict__ src, __bf16* __restrict__ dst,
                               int K, int N)
{
  size_t i = (size_t)blockIdx.x * blockDim.x + threadIdx.x;
  size_t total = (size_t)K * N;
  if (i >= total) return;
  int n = (int)(i / K), k = (int)(i % K);
  dst[i] = f2bf(src[(size_t)k * N + n]);
}

__global__ void concat_transpose_bf16(const float* __restrict__ Wih,
                                      const float* __restrict__ Whh,
                                      __bf16* __restrict__ dst)
{ // dst[NGATE][KCAT] = transpose of [W_ih;W_hh] ([KCAT,NGATE])
  size_t i = (size_t)blockIdx.x * blockDim.x + threadIdx.x;
  size_t total = (size_t)NGATE * KCAT;
  if (i >= total) return;
  int n = (int)(i / KCAT), k = (int)(i % KCAT);
  float v = (k < XIN) ? Wih[(size_t)k * NGATE + n] : Whh[(size_t)(k - XIN) * NGATE + n];
  dst[i] = f2bf(v);
}

__global__ void convert_bf16(const float* __restrict__ src, __bf16* __restrict__ dst, size_t n)
{
  size_t i = (size_t)blockIdx.x * blockDim.x + threadIdx.x;
  if (i < n) dst[i] = f2bf(src[i]);
}

__global__ void zero_words(uint32_t* __restrict__ p, size_t n)
{
  size_t i = (size_t)blockIdx.x * blockDim.x + threadIdx.x;
  if (i < n) p[i] = 0u;
}

// --- build xh = [a_embed | att_{t-1} | prodEmb[fp] | typeEmb[ft] | parent_h | h] bf16
__global__ void build_x(int t, const int* __restrict__ action_kind, const int* __restrict__ prod_ids,
                        const int* __restrict__ col_ids, const int* __restrict__ parent_prod,
                        const int* __restrict__ parent_type, const int* __restrict__ parent_time,
                        const int* __restrict__ tgt_lens, const float* __restrict__ prod_embed,
                        const float* __restrict__ type_embed, const float* __restrict__ colin,
                        const float* __restrict__ att, const __bf16* __restrict__ hbf,
                        const __bf16* __restrict__ hist, __bf16* __restrict__ xh)
{
  int b = blockIdx.x;
  int kp = 0, pp = 0, cp = 0, pvalid = 0;
  if (t > 0) {
    kp = action_kind[b * T_ + t - 1];
    pp = prod_ids[b * T_ + t - 1];
    cp = col_ids[b * T_ + t - 1];
    pvalid = ((t - 1) < tgt_lens[b]) ? 1 : 0;
  }
  int fprod = parent_prod[b * T_ + t];
  int ftype = parent_type[b * T_ + t];
  int ptime = parent_time[b * T_ + t];
  __bf16* out = xh + (size_t)b * KCAT;
  for (int pos = threadIdx.x; pos < KCAT; pos += blockDim.x) {
    __bf16 v = f2bf(0.0f);
    if (pos >= XIN) {
      v = hbf[(size_t)b * DRNN + (pos - XIN)];
    } else if (t != 0) {
      if (pos < DACT) {
        if (pvalid) {
          if (kp == 2) v = f2bf(colin[((size_t)b * C_ + cp) * DACT + pos]);
          else         v = f2bf(prod_embed[(size_t)(kp == 1 ? NPROD_ : pp) * DACT + pos]);
        }
      } else if (pos < DACT + DATT) {
        v = f2bf(att[(size_t)b * DATT + (pos - DACT)]);
      } else if (pos < DACT + DATT + DACT) {
        v = f2bf(prod_embed[(size_t)fprod * DACT + (pos - DACT - DATT)]);
      } else if (pos < DACT + DATT + DACT + DTYP) {
        v = f2bf(type_embed[(size_t)ftype * DTYP + (pos - DACT - DATT - DACT)]);
      } else {
        v = hist[((size_t)ptime * B_ + b) * DRNN + (pos - (DACT + DATT + DACT + DTYP))];
      }
    }
    out[pos] = v;
  }
}

// --- LSTM pointwise (torch gate order i,f,g,o) ---
__global__ void lstm_update(int t, const float* __restrict__ gates, float* __restrict__ h,
                            float* __restrict__ c, __bf16* __restrict__ hbf,
                            __bf16* __restrict__ hist)
{
  int b = blockIdx.x;
  for (int u = threadIdx.x; u < DRNN; u += blockDim.x) {
    const float* g = gates + (size_t)b * NGATE;
    float si = 1.f / (1.f + __expf(-g[u]));
    float sf = 1.f / (1.f + __expf(-g[DRNN + u]));
    float tg = tanhf(g[2 * DRNN + u]);
    float so = 1.f / (1.f + __expf(-g[3 * DRNN + u]));
    float cn = sf * c[(size_t)b * DRNN + u] + si * tg;
    float hn = so * tanhf(cn);
    c[(size_t)b * DRNN + u] = cn;
    h[(size_t)b * DRNN + u] = hn;
    hbf[(size_t)b * DRNN + u] = f2bf(hn);
    hist[((size_t)t * B_ + b) * DRNN + u] = f2bf(hn);
  }
}

// --- attention: softmax(q.enc) ctx; write [ctx|h] bf16. block=128 (=S_) ---
__global__ void attention_ctx(const float* __restrict__ q, const float* __restrict__ enc,
                              const int* __restrict__ enc_lens, const float* __restrict__ h,
                              __bf16* __restrict__ attcat)
{
  int b = blockIdx.x, tid = threadIdx.x;
  __shared__ float sp[S_];
  __shared__ float red[S_];
  const float* qb = q + (size_t)b * DENC;
  const float* eb = enc + (size_t)b * S_ * DENC;
  float sc = NEGV;
  {
    float d = 0.f;
    const float* er = eb + (size_t)tid * DENC;
    for (int k = 0; k < DENC; ++k) d += qb[k] * er[k];
    if (tid < enc_lens[b]) sc = d;
  }
  red[tid] = sc; __syncthreads();
  for (int s = 64; s > 0; s >>= 1) { if (tid < s) red[tid] = fmaxf(red[tid], red[tid + s]); __syncthreads(); }
  float m = red[0]; __syncthreads();
  float e = __expf(sc - m);
  red[tid] = e; __syncthreads();
  for (int s = 64; s > 0; s >>= 1) { if (tid < s) red[tid] += red[tid + s]; __syncthreads(); }
  float Z = red[0];
  sp[tid] = e / Z; __syncthreads();
  float c0 = 0.f, c1 = 0.f, c2 = 0.f, c3 = 0.f;
  for (int s = 0; s < S_; ++s) {
    float p = sp[s];
    const float* er = eb + (size_t)s * DENC;
    c0 += p * er[tid]; c1 += p * er[tid + 128]; c2 += p * er[tid + 256]; c3 += p * er[tid + 384];
  }
  __bf16* ob = attcat + (size_t)b * (DENC + DRNN);
  ob[tid] = f2bf(c0); ob[tid + 128] = f2bf(c1); ob[tid + 256] = f2bf(c2); ob[tid + 384] = f2bf(c3);
  const float* hb = h + (size_t)b * DRNN;
  ob[DENC + tid]       = f2bf(hb[tid]);
  ob[DENC + tid + 128] = f2bf(hb[tid + 128]);
  ob[DENC + tid + 256] = f2bf(hb[tid + 256]);
  ob[DENC + tid + 384] = f2bf(hb[tid + 384]);
}

// --- rule readout: logits = embq . prodEmb^T + b_lin; log_softmax; gather ---
__global__ void rule_logprob(int t, const float* __restrict__ embq,
                             const float* __restrict__ prod_embed, const float* __restrict__ b_lin,
                             const int* __restrict__ action_kind, const int* __restrict__ prod_ids,
                             float* __restrict__ lp_rule)
{
  int b = blockIdx.x, tid = threadIdx.x;  // block = 128, 98 rows used
  __shared__ float lg[128];
  __shared__ float red[128];
  float v = NEGV;
  if (tid <= NPROD_) {
    const float* e = embq + (size_t)b * DACT;
    const float* pr = prod_embed + (size_t)tid * DACT;
    float d = 0.f;
    for (int k = 0; k < DACT; ++k) d += e[k] * pr[k];
    v = d + b_lin[tid];
  }
  lg[tid] = v; red[tid] = v; __syncthreads();
  for (int s = 64; s > 0; s >>= 1) { if (tid < s) red[tid] = fmaxf(red[tid], red[tid + s]); __syncthreads(); }
  float m = red[0]; __syncthreads();
  red[tid] = (tid <= NPROD_) ? __expf(v - m) : 0.f; __syncthreads();
  for (int s = 64; s > 0; s >>= 1) { if (tid < s) red[tid] += red[tid + s]; __syncthreads(); }
  if (tid == 0) {
    int kind = action_kind[b * T_ + t];
    int pid  = prod_ids[b * T_ + t];
    int tr = (kind == 1) ? NPROD_ : pid;
    lp_rule[b] = lg[tr] - m - __logf(red[0]);
  }
}

// --- pointer net over columns + accumulate log prob. block = 64 (=C_) ---
__global__ void pointer_step(int t, const float* __restrict__ p, const float* __restrict__ col_enc,
                             const float* __restrict__ att, const float* __restrict__ w_mem,
                             const int* __restrict__ col_lens, const int* __restrict__ action_kind,
                             const int* __restrict__ col_ids, const int* __restrict__ tgt_lens,
                             const float* __restrict__ lp_rule, float* __restrict__ appear,
                             float* __restrict__ acc)
{
  int b = blockIdx.x, c = threadIdx.x;
  __shared__ float red[64];
  __shared__ float clp[64];
  const float* ab = att + (size_t)b * DATT;
  float gp = 0.f;
  for (int k = c; k < DATT; k += 64) gp += ab[k] * w_mem[k];
  red[c] = gp; __syncthreads();
  for (int s = 32; s > 0; s >>= 1) { if (c < s) red[c] += red[c + s]; __syncthreads(); }
  float gate = 1.f / (1.f + __expf(-red[0])); __syncthreads();

  const float* pb = p + (size_t)b * DATT;
  const float* ce = col_enc + ((size_t)b * C_ + c) * DENC;
  float wc = 0.f;
  for (int k = 0; k < DENC; ++k) wc += pb[k] * ce[k];
  float ap = appear[b * C_ + c];
  float w = wc * ap * gate + wc * (1.f - ap) * (1.f - gate);
  int clen = col_lens[b];
  float wt = (c < clen) ? w : NEGV;            // CTEMP == 1
  red[c] = wt; __syncthreads();
  for (int s = 32; s > 0; s >>= 1) { if (c < s) red[c] = fmaxf(red[c], red[c + s]); __syncthreads(); }
  float m = red[0]; __syncthreads();
  float e = __expf(wt - m);
  red[c] = e; __syncthreads();
  for (int s = 32; s > 0; s >>= 1) { if (c < s) red[c] += red[c + s]; __syncthreads(); }
  float lse = m + __logf(red[0]); __syncthreads();
  float lp_c = wt - lse;
  clp[c] = lp_c;
  red[c] = (c < clen) ? lp_c : 0.f; __syncthreads();
  for (int s = 32; s > 0; s >>= 1) { if (c < s) red[c] += red[c + s]; __syncthreads(); }

  int kind = action_kind[b * T_ + t];
  int cid  = col_ids[b * T_ + t];
  int valid = (t < tgt_lens[b]) ? 1 : 0;
  if (c == 0) {
    float lp_col = 0.8f * clp[cid] + 0.2f * red[0] / (float)clen;   // LS = 0.2
    float lp = (kind == 2) ? lp_col : lp_rule[b];
    if (valid) acc[b] += lp;
  }
  __syncthreads();
  if (c == cid && kind == 2 && valid) appear[b * C_ + c] = 1.0f;
}

__global__ void finalize_loss(const float* __restrict__ acc, float* __restrict__ out)
{
  __shared__ float red[B_];
  int tid = threadIdx.x;
  red[tid] = acc[tid]; __syncthreads();
  for (int s = 128; s > 0; s >>= 1) { if (tid < s) red[tid] += red[tid + s]; __syncthreads(); }
  if (tid == 0) out[0] = -red[0] / (float)B_;
}

// ---------------------------------------------------------------------------
extern "C" void kernel_launch(void* const* d_in, const int* in_sizes, int n_in,
                              void* d_out, int out_size, void* d_ws, size_t ws_size,
                              hipStream_t stream)
{
  (void)in_sizes; (void)n_in; (void)out_size; (void)ws_size;
  const float* enc         = (const float*)d_in[0];
  const float* col_enc     = (const float*)d_in[1];
  const int*   enc_lens    = (const int*)d_in[2];
  const int*   col_lens    = (const int*)d_in[3];
  const int*   action_kind = (const int*)d_in[4];
  const int*   tgt_lens    = (const int*)d_in[5];
  const int*   prod_ids    = (const int*)d_in[6];
  const int*   col_ids     = (const int*)d_in[7];
  const int*   parent_prod = (const int*)d_in[8];
  const int*   parent_type = (const int*)d_in[9];
  const int*   parent_time = (const int*)d_in[10];
  const float* prod_embed  = (const float*)d_in[11];
  const float* type_embed  = (const float*)d_in[12];
  const float* b_lin       = (const float*)d_in[13];
  const float* W_ih        = (const float*)d_in[14];
  const float* W_hh        = (const float*)d_in[15];
  const float* b_lstm      = (const float*)d_in[16];
  const float* W_att_q     = (const float*)d_in[17];
  const float* W_att_vec   = (const float*)d_in[18];
  const float* W_r1        = (const float*)d_in[19];
  const float* b_r1        = (const float*)d_in[20];
  const float* W_r2        = (const float*)d_in[21];
  const float* b_r2        = (const float*)d_in[22];
  const float* W_col_in    = (const float*)d_in[23];
  const float* W_ptr       = (const float*)d_in[24];
  const float* w_mem       = (const float*)d_in[25];

  char* ws = (char*)d_ws;
  size_t off = 0;
  auto alloc = [&](size_t bytes) -> char* {
    char* p = ws + off;
    off += (bytes + 255) & ~(size_t)255;
    return p;
  };
  // bf16 weights (pre-transposed to [N][K])
  __bf16* WcatT  = (__bf16*)alloc((size_t)NGATE * KCAT * 2);
  __bf16* WattqT = (__bf16*)alloc((size_t)DENC * DRNN * 2);
  __bf16* WattvT = (__bf16*)alloc((size_t)DATT * (DENC + DRNN) * 2);
  __bf16* Wr1T   = (__bf16*)alloc((size_t)(4 * DATT) * DATT * 2);
  __bf16* Wr2T   = (__bf16*)alloc((size_t)DACT * (4 * DATT) * 2);
  __bf16* WptrT  = (__bf16*)alloc((size_t)512 * DATT * 2);
  __bf16* WcolT  = (__bf16*)alloc((size_t)DACT * DENC * 2);
  __bf16* colenc_bf = (__bf16*)alloc((size_t)B_ * C_ * DENC * 2);
  // activations / state
  float*  colin  = (float*)alloc((size_t)B_ * C_ * DACT * 4);
  __bf16* xh     = (__bf16*)alloc((size_t)B_ * KCAT * 2);
  float*  gates  = (float*)alloc((size_t)B_ * NGATE * 4);
  float*  h_f32  = (float*)alloc((size_t)B_ * DRNN * 4);
  float*  c_f32  = (float*)alloc((size_t)B_ * DRNN * 4);
  __bf16* h_bf   = (__bf16*)alloc((size_t)B_ * DRNN * 2);
  float*  att_f32= (float*)alloc((size_t)B_ * DATT * 4);
  __bf16* att_bf = (__bf16*)alloc((size_t)B_ * DATT * 2);
  float*  q_f32  = (float*)alloc((size_t)B_ * DENC * 4);
  __bf16* attcat = (__bf16*)alloc((size_t)B_ * (DENC + DRNN) * 2);
  __bf16* r1_bf  = (__bf16*)alloc((size_t)B_ * 4 * DATT * 2);
  float*  embq   = (float*)alloc((size_t)B_ * DACT * 4);
  float*  p_f32  = (float*)alloc((size_t)B_ * 512 * 4);
  __bf16* hist_bf= (__bf16*)alloc((size_t)T_ * B_ * DRNN * 2);
  float*  appear = (float*)alloc((size_t)B_ * C_ * 4);
  float*  acc    = (float*)alloc((size_t)B_ * 4);
  float*  lp_rule= (float*)alloc((size_t)B_ * 4);

  auto gemm = [&](const __bf16* A, const __bf16* BT, const float* bias,
                  float* Cf, __bf16* Cb, int M, int N, int K, int act) {
    dim3 g(N / 64, M / 64);
    wmma_gemm<<<g, dim3(128), 0, stream>>>(A, BT, bias, Cf, Cb, M, N, K, act);
  };
  auto zero = [&](void* p, size_t words) {
    zero_words<<<(unsigned)((words + 255) / 256), 256, 0, stream>>>((uint32_t*)p, words);
  };

  // ---- weight prep (runs every call; deterministic, ~25MB, trivial cost) ----
  {
    size_t n = (size_t)NGATE * KCAT;
    concat_transpose_bf16<<<(unsigned)((n + 255) / 256), 256, 0, stream>>>(W_ih, W_hh, WcatT);
  }
  auto tconv = [&](const float* src, __bf16* dst, int K, int N) {
    size_t n = (size_t)K * N;
    transpose_bf16<<<(unsigned)((n + 255) / 256), 256, 0, stream>>>(src, dst, K, N);
  };
  tconv(W_att_q,   WattqT, DRNN, DENC);
  tconv(W_att_vec, WattvT, DENC + DRNN, DATT);
  tconv(W_r1,      Wr1T,   DATT, 4 * DATT);
  tconv(W_r2,      Wr2T,   4 * DATT, DACT);
  tconv(W_ptr,     WptrT,  DATT, 512);
  tconv(W_col_in,  WcolT,  DENC, DACT);
  {
    size_t n = (size_t)B_ * C_ * DENC;
    convert_bf16<<<(unsigned)((n + 255) / 256), 256, 0, stream>>>(col_enc, colenc_bf, n);
  }
  // col_in_embed = col_enc @ W_col_in  (M=16384, K=512, N=128) via WMMA
  gemm(colenc_bf, WcolT, nullptr, colin, nullptr, B_ * C_, DACT, DENC, 0);

  // ---- state init ----
  zero(h_f32,  (size_t)B_ * DRNN);
  zero(c_f32,  (size_t)B_ * DRNN);
  zero(h_bf,   (size_t)B_ * DRNN / 2);
  zero(att_f32,(size_t)B_ * DATT);
  zero(appear, (size_t)B_ * C_);
  zero(acc,    (size_t)B_);

  // ---- sequential decode ----
  for (int t = 0; t < T_; ++t) {
    build_x<<<B_, 256, 0, stream>>>(t, action_kind, prod_ids, col_ids, parent_prod,
                                    parent_type, parent_time, tgt_lens, prod_embed,
                                    type_embed, colin, att_f32, h_bf, hist_bf, xh);
    gemm(xh, WcatT, b_lstm, gates, nullptr, B_, NGATE, KCAT, 0);
    lstm_update<<<B_, 256, 0, stream>>>(t, gates, h_f32, c_f32, h_bf, hist_bf);
    gemm(h_bf, WattqT, nullptr, q_f32, nullptr, B_, DENC, DRNN, 0);
    attention_ctx<<<B_, 128, 0, stream>>>(q_f32, enc, enc_lens, h_f32, attcat);
    gemm(attcat, WattvT, nullptr, att_f32, att_bf, B_, DATT, DENC + DRNN, 1);
    gemm(att_bf, Wr1T, b_r1, nullptr, r1_bf, B_, 4 * DATT, DATT, 1);
    gemm(r1_bf, Wr2T, b_r2, embq, nullptr, B_, DACT, 4 * DATT, 0);
    rule_logprob<<<B_, 128, 0, stream>>>(t, embq, prod_embed, b_lin, action_kind,
                                         prod_ids, lp_rule);
    gemm(att_bf, WptrT, nullptr, p_f32, nullptr, B_, 512, DATT, 0);
    pointer_step<<<B_, 64, 0, stream>>>(t, p_f32, col_enc, att_f32, w_mem, col_lens,
                                        action_kind, col_ids, tgt_lens, lp_rule,
                                        appear, acc);
  }
  finalize_loss<<<1, B_, 0, stream>>>(acc, (float*)d_out);
}
